// DiffusionNavigator_901943132769
// MI455X (gfx1250) — compile-verified, matrix operand order checked by reading, not run on hardware
//
#include <hip/hip_runtime.h>
#include <math.h>

// ---------------------------------------------------------------------------
// CDNA5 / gfx1250 DiffusionNavigator: full 50-step DDPM sampling pipeline.
//
//  * All weight GEMMs -> v_wmma_f32_16x16x32_bf16, f32 accumulate. Packed
//    bf16 weights (~66MB) stay resident in the 192MB L2 across all 50 steps.
//  * Activations consumed as GEMM-A are produced directly in bf16: GEMM
//    inner loop is pure b128-load + WMMA, no conversions, no K tail.
//  * Wave = 16 cols x NT*16 rows, NT a template param (1/2/4): B (weight)
//    fragment loaded once per K-step feeds NT WMMAs, straight-line loop,
//    no per-tile branches (A buffers are padded to grid row coverage;
//    garbage tiles are computed but never stored).
//  * Attention (~2% of FLOPs) is wave-per-(b,h,q) VALU+LDS; q row staged
//    via global_load_async_to_lds_b128 (+ s_wait_asynccnt).
//  * lan (lang_self stack) is timestep-invariant -> computed once, not 50x.
// ---------------------------------------------------------------------------

typedef __attribute__((ext_vector_type(16))) __bf16 v16bf;
typedef __attribute__((ext_vector_type(8)))  float  v8f;

static __device__ __forceinline__ __bf16 to_bf16(float f) { return (__bf16)f; }

// ============================ device RNG ===================================
__device__ __forceinline__ unsigned mixu(unsigned x) {
  x ^= x >> 16; x *= 0x7feb352du; x ^= x >> 15; x *= 0x846ca68bu; x ^= x >> 16;
  return x;
}
__device__ __forceinline__ float rng_normal(unsigned seed, unsigned i) {
  unsigned a = mixu(seed ^ (i * 0x85ebca6bu) ^ 0x165667b1u);
  unsigned b = mixu(a + 0x9e3779b9u);
  float u1 = ((a >> 8) + 1u) * (1.0f / 16777216.0f);
  float u2 = (b >> 8) * (1.0f / 16777216.0f);
  return sqrtf(-2.0f * logf(u1)) * cosf(6.2831853071f * u2);
}

// ============================ packing ======================================
__global__ void pack_wt_kernel(const float* __restrict__ W,
                               __bf16* __restrict__ Wt,
                               int K, int N, int Kp) {
  int k = blockIdx.x * blockDim.x + threadIdx.x;
  int n = blockIdx.y;
  if (k >= Kp || n >= N) return;
  float v = (k < K) ? W[(size_t)k * N + n] : 0.0f;
  Wt[(size_t)n * Kp + k] = to_bf16(v);
}

__global__ void pack_act_kernel(const float* __restrict__ x,
                                __bf16* __restrict__ y, int K, int Kp) {
  int k = blockIdx.x * blockDim.x + threadIdx.x;
  int row = blockIdx.y;
  if (k >= Kp) return;
  y[(size_t)row * Kp + k] = (k < K) ? to_bf16(x[(size_t)row * K + k]) : to_bf16(0.f);
}

// ============================ WMMA GEMM ====================================
// C[M,N] = act( A @ W + bias (+ residual) ).  Ab bf16 [Mpad,Kp], Wt [N,Kp].
// One wave = 16 cols x NT*16 rows. Straight-line inner loop, no branches.
template <int NT>
__global__ __launch_bounds__(128) void gemm_bf16_wmma(
    const __bf16* __restrict__ Ab,
    const __bf16* __restrict__ Wt,
    const float* __restrict__ bias,
    const float* __restrict__ residual,
    float* __restrict__ Cf,
    __bf16* __restrict__ Cb,
    int M, int N, int Kp, int act) {
  const int lane = threadIdx.x & 31;
  const int wave = threadIdx.x >> 5;
  const int n0 = (blockIdx.x * 4 + wave) * 16;
  const int m0 = blockIdx.y * (NT * 16);
  if (n0 >= N) return;                  // wave-uniform

  const int r   = lane & 15;
  const int col = n0 + r;
  const int kb  = (lane < 16) ? 0 : 8;
  const __bf16* bp = Wt + (size_t)col * Kp + kb;
  const __bf16* ap = Ab + (size_t)(m0 + r) * Kp + kb;   // tiles 16*Kp apart

  v8f acc[NT] = {};
  for (int k0 = 0; k0 < Kp; k0 += 32) {
    union { int4 i4[2]; v16bf v; } bu;
    bu.i4[0] = *(const int4*)(bp + k0);
    bu.i4[1] = *(const int4*)(bp + k0 + 16);
    #pragma unroll
    for (int tl = 0; tl < NT; ++tl) {
      union { int4 i4[2]; v16bf v; } au;
      au.i4[0] = *(const int4*)(ap + (size_t)tl * 16 * Kp + k0);
      au.i4[1] = *(const int4*)(ap + (size_t)tl * 16 * Kp + k0 + 16);
      acc[tl] = __builtin_amdgcn_wmma_f32_16x16x32_bf16(
          false, au.v, false, bu.v, (short)0, acc[tl], false, false);
    }
  }

  const float bv = bias ? bias[col] : 0.0f;
  #pragma unroll
  for (int tl = 0; tl < NT; ++tl) {
    #pragma unroll
    for (int j = 0; j < 8; ++j) {
      int m = m0 + tl * 16 + j + ((lane < 16) ? 0 : 8);
      if (m < M) {
        float v = acc[tl][j] + bv;
        if (residual) v += residual[(size_t)m * N + col];
        if (act == 1) v = fmaxf(v, 0.0f);
        if (Cf) Cf[(size_t)m * N + col] = v;
        if (Cb) Cb[(size_t)m * N + col] = to_bf16(v);
      }
    }
  }
}

// ============================ LayerNorm (+cond) ============================
__global__ __launch_bounds__(128) void layernorm_kernel(
    const float* __restrict__ x, const float* __restrict__ g,
    const float* __restrict__ bt, const float* __restrict__ cond,
    __bf16* __restrict__ y, int rows, int L) {
  int row = blockIdx.x * 4 + (threadIdx.x >> 5);
  int lane = threadIdx.x & 31;
  if (row >= rows) return;
  const float* xp = x + (size_t)row * 512;
  float vals[16];
  float s = 0.f;
  #pragma unroll
  for (int c = 0; c < 16; ++c) { float v = xp[lane + 32 * c]; vals[c] = v; s += v; }
  #pragma unroll
  for (int o = 16; o; o >>= 1) s += __shfl_xor(s, o, 32);
  float mean = s * (1.0f / 512.0f);
  float vs = 0.f;
  #pragma unroll
  for (int c = 0; c < 16; ++c) { float d = vals[c] - mean; vs += d * d; }
  #pragma unroll
  for (int o = 16; o; o >>= 1) vs += __shfl_xor(vs, o, 32);
  float rstd = rsqrtf(vs * (1.0f / 512.0f) + 1e-5f);
  int b = row / L;
  __bf16* yp = y + (size_t)row * 512;
  #pragma unroll
  for (int c = 0; c < 16; ++c) {
    int d = lane + 32 * c;
    float v = (vals[c] - mean) * rstd * g[d] + bt[d];
    if (cond) v += cond[b * 512 + d];
    yp[d] = to_bf16(v);
  }
}

// ============================ attention ====================================
__global__ __launch_bounds__(32) void attention_kernel(
    const float* __restrict__ Q, const float* __restrict__ K,
    const float* __restrict__ V, __bf16* __restrict__ O,
    const int* __restrict__ mask_tok, int Lq, int Lk, float scale) {
  const int qi = blockIdx.x, h = blockIdx.y, b = blockIdx.z;
  const int lane = threadIdx.x;
  __shared__ alignas(16) float sq[64];
  __shared__ float prob[128];

  // Stage q row (256B) memory -> LDS via gfx1250 async DMA, lanes 0..15.
  const float* qp = Q + ((size_t)(b * Lq + qi)) * 512 + h * 64;
  if (lane < 16) {
    unsigned lds_off = (unsigned)(size_t)(&sq[0]) + (unsigned)lane * 16u;
    const float* ga = qp + lane * 4;
    asm volatile("global_load_async_to_lds_b128 %0, %1, off"
                 :: "v"(lds_off), "v"(ga) : "memory");
  }
  asm volatile("s_wait_asynccnt 0" ::: "memory");
  __syncthreads();

  float sc[4];
  float mx = -3.4e38f;
  #pragma unroll
  for (int c = 0; c < 4; ++c) {
    int j = lane + 32 * c;
    float s = -3.4e38f;
    if (j < Lk) {
      const float* kp = K + ((size_t)(b * Lk + j)) * 512 + h * 64;
      float d = 0.f;
      #pragma unroll 8
      for (int e = 0; e < 64; ++e) d += sq[e] * kp[e];
      s = d * scale;
      if (mask_tok && mask_tok[b * Lk + j] == 0) s = -1e9f;
    }
    sc[c] = s;
    mx = fmaxf(mx, s);
  }
  #pragma unroll
  for (int o = 16; o; o >>= 1) mx = fmaxf(mx, __shfl_xor(mx, o, 32));
  float sum = 0.f;
  #pragma unroll
  for (int c = 0; c < 4; ++c) {
    int j = lane + 32 * c;
    float e = (j < Lk) ? expf(sc[c] - mx) : 0.f;
    sc[c] = e;
    sum += e;
  }
  #pragma unroll
  for (int o = 16; o; o >>= 1) sum += __shfl_xor(sum, o, 32);
  float inv = 1.0f / sum;
  #pragma unroll
  for (int c = 0; c < 4; ++c) { int j = lane + 32 * c; if (j < 128) prob[j] = sc[c] * inv; }
  __syncthreads();

  float a0 = 0.f, a1 = 0.f;
  for (int j = 0; j < Lk; ++j) {
    float p = prob[j];
    const float* vp = V + ((size_t)(b * Lk + j)) * 512 + h * 64;
    a0 += p * vp[lane];
    a1 += p * vp[lane + 32];
  }
  __bf16* op = O + ((size_t)(b * Lq + qi)) * 512 + h * 64;
  op[lane] = to_bf16(a0);
  op[lane + 32] = to_bf16(a1);
}

// ============================ embeddings / PE ==============================
__global__ void embed_instr_kernel(const int* __restrict__ tok,
                                   const float* __restrict__ emb,
                                   float* __restrict__ out, int LI) {
  int d = blockIdx.x * 256 + threadIdx.x;
  int l = blockIdx.y, b = blockIdx.z;
  int t = tok[b * LI + l];
  float div = expf((float)(d & ~1) * (-9.2103403720f / 512.0f));
  float pe = (d & 1) ? cosf(l * div) : sinf(l * div);
  out[((size_t)(b * LI + l)) * 512 + d] = emb[(size_t)t * 512 + d] + pe;
}

__global__ void add_pe_kernel(const float* __restrict__ x,
                              float* __restrict__ out, int L) {
  int d = blockIdx.x * 256 + threadIdx.x;
  int l = blockIdx.y, b = blockIdx.z;
  float div = expf((float)(d & ~1) * (-9.2103403720f / 512.0f));
  float pe = (d & 1) ? cosf(l * div) : sinf(l * div);
  size_t o = ((size_t)(b * L + l)) * 512 + d;
  out[o] = x[o] + pe;
}

__global__ void sinusoidal_kernel(const int* __restrict__ times, float tconst,
                                  __bf16* __restrict__ out) {
  int d = blockIdx.x * 256 + threadIdx.x;
  int b = blockIdx.y;
  float t = times ? (float)times[b] : tconst;
  int i = (d < 256) ? d : d - 256;
  float fr = expf((float)i * (-9.2103403720f / 255.0f));
  out[b * 512 + d] = to_bf16((d < 256) ? sinf(t * fr) : cosf(t * fr));
}

// ============================ DDPM update ==================================
__global__ void init_noise_kernel(float* __restrict__ x, int n) {
  int i = blockIdx.x * 256 + threadIdx.x;
  if (i < n) x[i] = rng_normal(0x2545F491u, (unsigned)i);
}

__global__ void ddpm_step_kernel(float* __restrict__ x, const float* __restrict__ eps,
                                 float s1mat, float inv_sat, float c1, float c2,
                                 float sigma, int t, int n) {
  int i = blockIdx.x * 256 + threadIdx.x;
  if (i >= n) return;
  float xv = x[i];
  float x0 = (xv - s1mat * eps[i]) * inv_sat;
  x0 = fminf(fmaxf(x0, -1.0f), 1.0f);
  float prev = c1 * x0 + c2 * xv;
  float z = rng_normal((unsigned)(t + 1) * 0x9E3779B9u, (unsigned)i);
  x[i] = prev + sigma * z;
}

// ============================ finalize =====================================
__global__ __launch_bounds__(32) void finalize_kernel(const float* __restrict__ den,
                                                      const float* __restrict__ act_emb,
                                                      float* __restrict__ out) {
  int l = blockIdx.x, b = blockIdx.y, lane = threadIdx.x;
  const float* xp = den + ((size_t)(b * 16 + l)) * 512;
  for (int e = lane; e < 512; e += 32) out[((size_t)(b * 16 + l)) * 512 + e] = xp[e];
  float ds[4];
  #pragma unroll
  for (int a = 0; a < 4; ++a) {
    const float* ap = act_emb + a * 512;
    float s = 0.f;
    for (int e = lane; e < 512; e += 32) { float d = xp[e] - ap[e]; s += d * d; }
    #pragma unroll
    for (int o = 16; o; o >>= 1) s += __shfl_xor(s, o, 32);
    ds[a] = s;
  }
  if (lane == 0) {
    int best = 0; float bv = ds[0];
    #pragma unroll
    for (int a = 1; a < 4; ++a) if (ds[a] < bv) { bv = ds[a]; best = a; }
    out[2 * 16 * 512 + b * 16 + l] = (float)best;
  }
}

// ============================ host orchestration ===========================

namespace {
struct PLin { const float* w; const float* b; __bf16* wt; int K, N, Kp; };
struct PLayer {
  PLin wq, wk, wv, wo, f1, f2;
  const float* ln1g; const float* ln1b;
  const float* ln2g; const float* ln2b;
  int ffn;
};
struct Bump {
  char* base; size_t off;
  void* take(size_t bytes) {
    void* p = base + off;
    off += (bytes + 255) & ~(size_t)255;
    return p;
  }
};
inline int ru32(int k) { return (k + 31) & ~31; }
}  // namespace

extern "C" void kernel_launch(void* const* d_in, const int* in_sizes, int n_in,
                              void* d_out, int out_size, void* d_ws, size_t ws_size,
                              hipStream_t stream) {
  (void)in_sizes; (void)n_in; (void)out_size; (void)ws_size;

  const int D = 512, NH = 8, B = 2, LI = 77, NT = 64, AL = 16, T = 50;

  Bump ws{(char*)d_ws, 0};

  // -------- walk input leaves (depth-first insertion order) --------
  int ii = 0;
  auto NXT = [&]() { return (const float*)d_in[ii++]; };

  const float* instr_emb  = NXT();  // [2000,512]
  const float* action_emb = NXT();  // [4,512]

  auto read_lin = [&](int K, int N) {
    PLin L; L.w = NXT(); L.b = NXT(); L.K = K; L.N = N; L.Kp = ru32(K);
    L.wt = (__bf16*)ws.take((size_t)N * L.Kp * sizeof(__bf16));
    return L;
  };
  auto read_mat = [&](int K, int N) {
    PLin L; L.w = NXT(); L.b = nullptr; L.K = K; L.N = N; L.Kp = ru32(K);
    L.wt = (__bf16*)ws.take((size_t)N * L.Kp * sizeof(__bf16));
    return L;
  };
  auto read_layer = [&](int ffn) {
    PLayer L;
    L.wq = read_mat(D, D); L.wk = read_mat(D, D);
    L.wv = read_mat(D, D); L.wo = read_mat(D, D);
    L.ln1g = NXT(); L.ln1b = NXT();
    L.ffn = ffn;
    if (ffn) {
      L.f1 = read_lin(D, 4 * D); L.f2 = read_lin(4 * D, D);
      L.ln2g = NXT(); L.ln2b = NXT();
    } else {
      L.ln2g = L.ln2b = nullptr;
      L.f1 = PLin{}; L.f2 = PLin{};
    }
    return L;
  };

  PLin rgb_lin   = read_lin(16, D);
  PLin depth_lin = read_lin(16, D);
  PLin time1 = read_lin(D, D), time2 = read_lin(D, D);
  PLin seq1  = read_lin(D, D), seq2  = read_lin(D, D);
  PLayer lang[2]  = {read_layer(1), read_layer(1)};
  PLayer obsl[2]  = {read_layer(1), read_layer(1)};
  PLayer vll[2]   = {read_layer(1), read_layer(1)};
  PLayer traj     = read_layer(0);
  PLayer csl[2]   = {read_layer(1), read_layer(1)};
  PLayer fsl[2]   = {read_layer(1), read_layer(1)};
  PLin np1 = read_lin(D, D), np2 = read_lin(D, D);

  const int*   instruction = (const int*)d_in[ii++];    // [2,77]
  const float* rgb_f       = NXT();                     // [2,64,16]
  const float* depth_f     = NXT();                     // [2,64,16]
  const int*   seq_t       = (const int*)d_in[ii++];    // [2]

  // -------- activation buffers --------
  // bf16 GEMM-A buffers are padded to the row coverage of the GEMM grid
  // (192 rows for the 154-row tensors, 16 rows for the [2,512] MLP inputs):
  // out-of-range tiles are computed (garbage) but never stored.
  auto fbuf = [&](size_t n) { return (float*)ws.take(n * sizeof(float)); };
  auto bbuf = [&](size_t n) { return (__bf16*)ws.take(n * sizeof(__bf16)); };
  const size_t RMAX  = (size_t)B * LI * D;   // 154*512 (f32 tensors)
  const size_t RPAD  = (size_t)192 * D;      // padded rows for bf16 GEMM-A
  // f32 (residual / LN / attention / ddpm consumers)
  float* t_q   = fbuf(RMAX);
  float* t_k   = fbuf(RMAX);
  float* t_v   = fbuf(RMAX);
  float* t_x2  = fbuf(RMAX);
  float* t_pp  = fbuf(RMAX);
  float* instr_pe = fbuf(RMAX);
  float* rgb   = fbuf((size_t)B * NT * D);
  float* obs   = fbuf((size_t)B * NT * D);
  float* seq_emb = fbuf((size_t)B * D);
  float* t_emb   = fbuf((size_t)B * D);
  float* xact  = fbuf((size_t)B * AL * D);
  float* actpe = fbuf((size_t)B * AL * D);
  float* bt1   = fbuf((size_t)B * AL * D);
  float* bt2   = fbuf((size_t)B * AL * D);
  float* epsb  = fbuf((size_t)B * AL * D);
  // bf16 (GEMM-A consumers; padded)
  __bf16* t_hb  = bbuf(RPAD);
  __bf16* t_aob = bbuf(RPAD);
  __bf16* t_ffb = bbuf((size_t)192 * 4 * D);
  __bf16* sin2b = bbuf((size_t)16 * D);
  __bf16* mhb   = bbuf((size_t)16 * D);
  __bf16* rgbf_b = bbuf((size_t)B * NT * 32);
  __bf16* depf_b = bbuf((size_t)B * NT * 32);
  __bf16* dep_b  = bbuf((size_t)B * NT * D);
  __bf16* lan_b  = bbuf(RPAD);
  __bf16* ctx_b  = bbuf((size_t)B * NT * D);
  __bf16* bt3b   = bbuf((size_t)B * AL * D);
  __bf16* tnpb   = bbuf((size_t)B * AL * D);

  // -------- launch helpers --------
  auto pack = [&](PLin& L) {
    dim3 g((L.Kp + 63) / 64, L.N);
    pack_wt_kernel<<<g, 64, 0, stream>>>(L.w, L.wt, L.K, L.N, L.Kp);
  };
  auto gemm = [&](const __bf16* Ab, PLin& L, const float* res, float* Cf, __bf16* Cb,
                  int M, int act) {
    int tiles = (M + 15) >> 4;
    int nt = (tiles >= 4) ? 4 : ((tiles >= 2) ? 2 : 1);
    dim3 g((L.N + 63) / 64, (tiles + nt - 1) / nt);
    if (nt == 4)
      gemm_bf16_wmma<4><<<g, 128, 0, stream>>>(Ab, L.wt, L.b, res, Cf, Cb, M, L.N, L.Kp, act);
    else if (nt == 2)
      gemm_bf16_wmma<2><<<g, 128, 0, stream>>>(Ab, L.wt, L.b, res, Cf, Cb, M, L.N, L.Kp, act);
    else
      gemm_bf16_wmma<1><<<g, 128, 0, stream>>>(Ab, L.wt, L.b, res, Cf, Cb, M, L.N, L.Kp, act);
  };
  auto ln = [&](const float* x, const float* g, const float* b2, const float* cond,
                __bf16* y, int rows, int L) {
    layernorm_kernel<<<dim3((rows + 3) / 4), 128, 0, stream>>>(x, g, b2, cond, y, rows, L);
  };
  auto attn = [&](const float* Q, const float* K, const float* V, __bf16* O,
                  const int* mask, int Lq, int Lk) {
    attention_kernel<<<dim3(Lq, NH, B), 32, 0, stream>>>(Q, K, V, O, mask, Lq, Lk, 0.125f);
  };
  auto layer_apply = [&](PLayer& L, const float* xin, float* xoutF, __bf16* xoutB,
                         const __bf16* kvb, int Lk, const float* cond,
                         const int* mask, int Lq) {
    const int rows = B * Lq, rowsk = B * Lk;
    ln(xin, L.ln1g, L.ln1b, cond, t_hb, rows, Lq);
    gemm(t_hb, L.wq, nullptr, t_q, nullptr, rows, 0);
    const __bf16* src = kvb ? kvb : t_hb;
    gemm(src, L.wk, nullptr, t_k, nullptr, rowsk, 0);
    gemm(src, L.wv, nullptr, t_v, nullptr, rowsk, 0);
    attn(t_q, t_k, t_v, t_aob, mask, Lq, Lk);
    if (L.ffn) {
      gemm(t_aob, L.wo, xin, t_x2, nullptr, rows, 0);      // + residual xin
      ln(t_x2, L.ln2g, L.ln2b, nullptr, t_hb, rows, Lq);
      gemm(t_hb, L.f1, nullptr, nullptr, t_ffb, rows, 1);  // ReLU, bf16 out
      gemm(t_ffb, L.f2, t_x2, xoutF, xoutB, rows, 0);      // + residual x2
    } else {
      gemm(t_aob, L.wo, xin, xoutF, xoutB, rows, 0);
    }
  };
  auto stack2 = [&](PLayer* Ls, const float* xin, float* xoutF, __bf16* xoutB,
                    const __bf16* kvb, int Lk, const float* cond,
                    const int* mask, int Lq) {
    layer_apply(Ls[0], xin, t_pp, nullptr, kvb, Lk, cond, mask, Lq);
    layer_apply(Ls[1], t_pp, xoutF, xoutB, kvb, Lk, cond, mask, Lq);
  };

  // -------- pack all weights (bf16, transposed) --------
  pack(rgb_lin); pack(depth_lin);
  pack(time1); pack(time2); pack(seq1); pack(seq2); pack(np1); pack(np2);
  for (int i = 0; i < 2; ++i) {
    PLayer* Ls[5] = {&lang[i], &obsl[i], &vll[i], &csl[i], &fsl[i]};
    for (PLayer* L : Ls) {
      pack(L->wq); pack(L->wk); pack(L->wv); pack(L->wo);
      pack(L->f1); pack(L->f2);
    }
  }
  pack(traj.wq); pack(traj.wk); pack(traj.wv); pack(traj.wo);

  // -------- timestep-invariant prologue --------
  embed_instr_kernel<<<dim3(2, LI, B), 256, 0, stream>>>(instruction, instr_emb, instr_pe, LI);

  sinusoidal_kernel<<<dim3(2, B), 256, 0, stream>>>(seq_t, 0.f, sin2b);
  gemm(sin2b, seq1, nullptr, nullptr, mhb, B, 1);
  gemm(mhb, seq2, nullptr, seq_emb, nullptr, B, 0);

  pack_act_kernel<<<dim3(1, B * NT), 64, 0, stream>>>(rgb_f, rgbf_b, 16, 32);
  pack_act_kernel<<<dim3(1, B * NT), 64, 0, stream>>>(depth_f, depf_b, 16, 32);
  gemm(rgbf_b, rgb_lin, nullptr, rgb, nullptr, B * NT, 0);
  gemm(depf_b, depth_lin, nullptr, nullptr, dep_b, B * NT, 0);

  // lan depends only on instr/seq: hoisted out of the 50-step loop.
  stack2(lang, instr_pe, nullptr, lan_b, nullptr, LI, seq_emb, instruction, LI);

  const int NACT = B * AL * D;   // 16384
  init_noise_kernel<<<(NACT + 255) / 256, 256, 0, stream>>>(xact, NACT);

  // -------- DDPM scheduler constants (host, double precision) --------
  double acum[T];
  {
    double prev = 1.0;
    const double s0 = sqrt(1e-4), s1 = sqrt(0.02);
    for (int i = 0; i < T; ++i) {
      double bq = s0 + (s1 - s0) * ((double)i / (double)(T - 1));
      prev *= (1.0 - bq * bq);
      acum[i] = prev;
    }
  }

  // -------- 50-step denoising loop --------
  for (int t = T - 1; t >= 0; --t) {
    sinusoidal_kernel<<<dim3(2, B), 256, 0, stream>>>(nullptr, (float)t, sin2b);
    gemm(sin2b, time1, nullptr, nullptr, mhb, B, 1);
    gemm(mhb, time2, nullptr, t_emb, nullptr, B, 0);

    stack2(obsl, rgb, obs, nullptr, dep_b, NT, t_emb, nullptr, NT);
    stack2(vll, obs, nullptr, ctx_b, lan_b, LI, nullptr, instruction, NT);

    add_pe_kernel<<<dim3(2, AL, B), 256, 0, stream>>>(xact, actpe, AL);
    layer_apply(traj, actpe, bt1, nullptr, lan_b, LI, nullptr, instruction, AL);
    stack2(csl, bt1, bt2, nullptr, ctx_b, NT, t_emb, nullptr, AL);
    stack2(fsl, bt2, nullptr, bt3b, nullptr, AL, t_emb, nullptr, AL);

    gemm(bt3b, np1, nullptr, nullptr, tnpb, B * AL, 1);
    gemm(tnpb, np2, nullptr, epsb, nullptr, B * AL, 0);

    double a_t = acum[t];
    double a_prev = (t > 0) ? acum[t - 1] : 1.0;
    double ca = a_t / a_prev, cb = 1.0 - ca;
    float c1 = (float)(sqrt(a_prev) * cb / (1.0 - a_t));
    float c2 = (float)(sqrt(ca) * (1.0 - a_prev) / (1.0 - a_t));
    double var = (1.0 - a_prev) / (1.0 - a_t) * cb;
    if (var < 1e-20) var = 1e-20;
    float sigma = (t > 0) ? (float)sqrt(var) : 0.0f;
    float s1m = (float)sqrt(1.0 - a_t);
    float isa = (float)(1.0 / sqrt(a_t));
    ddpm_step_kernel<<<(NACT + 255) / 256, 256, 0, stream>>>(
        xact, epsb, s1m, isa, c1, c2, sigma, t, NACT);
  }

  // -------- output: denoised + nearest-action argmin --------
  finalize_kernel<<<dim3(AL, B), 32, 0, stream>>>(xact, action_emb, (float*)d_out);
}
